// GNS_90013924590495
// MI455X (gfx1250) — matrix operand order repaired and checked by reading, not compiled, not run on hardware
//
#include <hip/hip_runtime.h>
#include <cstdint>
#include <cstddef>

// ---------------- problem constants (match reference) ----------------
#define NN 16384     // nodes
#define NE 65536     // edges
#define HD 128       // hidden dim
#define RWALL 0.015f

typedef __bf16 bf16;
typedef unsigned int uint32;
typedef __attribute__((ext_vector_type(16))) __bf16 bf16x16;
typedef __attribute__((ext_vector_type(8)))  __bf16 bf16x8;
typedef __attribute__((ext_vector_type(8)))  float  f32x8;

// =====================================================================
// Generic 128-wide GEMM:  Y[M x 128] = act(X[M x K] @ W[K x 128] + b)
// A: bf16 row-major (or fused gather of [edges | nodes[recv] | nodes[send]])
// B: bf16 weights pre-transposed: Wt[n][k], n=0..127, k=0..K-1 (K % 32 == 0)
// Each wave32 computes a (RT*16) x 128 output tile with
// v_wmma_f32_16x16x32_bf16. K and gather-mode are compile-time, so the
// whole inner loop is saddr+voffset+imm loads and WMMAs: no address math.
// =====================================================================
struct GemmP {
  const bf16* X; int ldx;             // plain A source
  const bf16* edges;                  // gather mode sources
  const bf16* xn;                     // node state N x 256 (cols 128..255 = node emb)
  const int*  nbr;                    // [E][2] = (recv, send)
  const bf16* Wt;                     // 128 x K bf16
  const float* bias;                  // 128 fp32
  int relu;
  bf16* Ybf; int ldy; int ycol;       // optional bf16 output
  float* Yf;                          // optional fp32 output (M x 128)
};

__device__ __forceinline__ bf16x16 join16(bf16x8 lo, bf16x8 hi) {
  return __builtin_shufflevector(lo, hi, 0,1,2,3,4,5,6,7,8,9,10,11,12,13,14,15);
}

template <int RT, int K, bool GATHER>
__global__ __launch_bounds__(128) void gemm128(GemmP p) {
  const int lane  = threadIdx.x & 31;
  const int wave  = threadIdx.x >> 5;
  const int rowBase = (blockIdx.x * 4 + wave) * (RT * 16);
  const int lrow  = lane & 15;      // A row / B col / C col within 16-tile
  const int lhalf = lane >> 4;      // 0: lanes 0-15, 1: lanes 16-31
  const uint32 laneK = (uint32)lhalf * 16u;   // byte offset of this half's k-chunk

  // --- per-tile A voffsets (32-bit); bases are uniform kernarg pointers
  uint32 aV[RT], eV[RT], rV[RT], sV[RT];
#pragma unroll
  for (int j = 0; j < RT; ++j) {
    const int r = rowBase + j * 16 + lrow;
    if (GATHER) {
      eV[j] = (uint32)r * (HD * 2u) + laneK;
      rV[j] = (uint32)p.nbr[2 * r + 0] * 512u + 256u + laneK;
      sV[j] = (uint32)p.nbr[2 * r + 1] * 512u + 256u + laneK;
      aV[j] = 0;
    } else {
      aV[j] = (uint32)r * (uint32)p.ldx * 2u + laneK;
      eV[j] = rV[j] = sV[j] = 0;
    }
  }

  f32x8 acc[RT][8] = {};   // (RT*16) rows x 128 cols, fp32 accum

  const char* wB = (const char*)p.Wt;
  const char* xB = (const char*)p.X;
  const char* eB = (const char*)p.edges;
  const char* nB = (const char*)p.xn;
  const uint32 laneB = (uint32)lrow * (K * 2u) + (uint32)lhalf * 16u;

#pragma unroll
  for (int kb = 0; kb < K; kb += 32) {
    bf16x16 A[RT];
#pragma unroll
    for (int j = 0; j < RT; ++j) {
      if (GATHER) {
        // compile-time region select: 0 = edges, 1 = nodes[recv], 2 = nodes[send]
        const int reg0 = kb >> 7;
        const int reg1 = (kb + 16) >> 7;
        const char* b0 = (reg0 == 0) ? eB : nB;
        const char* b1 = (reg1 == 0) ? eB : nB;
        const uint32 v0 = (reg0 == 0) ? eV[j] : (reg0 == 1) ? rV[j] : sV[j];
        const uint32 v1 = (reg1 == 0) ? eV[j] : (reg1 == 1) ? rV[j] : sV[j];
        bf16x8 lo = *(const bf16x8*)(b0 + v0 + (uint32)((kb      - reg0 * 128) * 2));
        bf16x8 hi = *(const bf16x8*)(b1 + v1 + (uint32)((kb + 16 - reg1 * 128) * 2));
        A[j] = join16(lo, hi);
      } else {
        bf16x8 lo = *(const bf16x8*)(xB + aV[j] + (uint32)(kb * 2));
        bf16x8 hi = *(const bf16x8*)(xB + aV[j] + (uint32)(kb * 2 + 32));
        A[j] = join16(lo, hi);
      }
    }
#pragma unroll
    for (int t = 0; t < 8; ++t) {
      bf16x16 B = join16(*(const bf16x8*)(wB + laneB + (uint32)(kb * 2 + t * (K * 32))),
                         *(const bf16x8*)(wB + laneB + (uint32)(kb * 2 + 32 + t * (K * 32))));
#pragma unroll
      for (int j = 0; j < RT; ++j)
        acc[j][t] = __builtin_amdgcn_wmma_f32_16x16x32_bf16(
            false, A[j], false, B, (short)0, acc[j][t], false, false);
    }
  }

  // Epilogue: bias (+ReLU) ; C layout: lane col = t*16+lrow, row = v + lhalf*8
  float bv[8];
#pragma unroll
  for (int t = 0; t < 8; ++t) bv[t] = p.bias[t * 16 + lrow];

  char* ybf = (char*)p.Ybf;
  char* yf  = (char*)p.Yf;
#pragma unroll
  for (int j = 0; j < RT; ++j) {
#pragma unroll
    for (int t = 0; t < 8; ++t) {
      const int col = t * 16 + lrow;
#pragma unroll
      for (int v = 0; v < 8; ++v) {
        const int row = rowBase + j * 16 + lhalf * 8 + v;
        float y = acc[j][t][v] + bv[t];
        if (p.relu) y = fmaxf(y, 0.0f);
        if (p.Ybf) {
          uint32 o = ((uint32)row * (uint32)p.ldy + (uint32)(p.ycol + col)) * 2u;
          *(bf16*)(ybf + o) = (bf16)y;
        }
        if (p.Yf) {
          uint32 o = ((uint32)row * (uint32)HD + (uint32)col) * 4u;
          *(float*)(yf + o) = y;
        }
      }
    }
  }
}

// ============== LayerNorm: one wave per 128-wide row, fp32 in, bf16 out ====
__global__ void ln_kernel(const float* __restrict__ Y, const float* __restrict__ g,
                          const float* __restrict__ b, bf16* __restrict__ dst,
                          int ldy, int ycol, int M) {
  int gid = blockIdx.x * blockDim.x + threadIdx.x;
  int row = gid >> 5;
  int lane = gid & 31;
  if (row >= M) return;
  float4 v = ((const float4*)(Y + (size_t)row * HD))[lane];
  float s = v.x + v.y + v.z + v.w;
  float q = v.x*v.x + v.y*v.y + v.z*v.z + v.w*v.w;
#pragma unroll
  for (int off = 16; off > 0; off >>= 1) {
    s += __shfl_xor(s, off, 32);
    q += __shfl_xor(q, off, 32);
  }
  float mean = s * (1.0f / 128.0f);
  float var  = q * (1.0f / 128.0f) - mean * mean;
  float rstd = rsqrtf(var + 1e-5f);
  bf16* o = dst + (size_t)row * ldy + ycol + lane * 4;
  float vv[4] = {v.x, v.y, v.z, v.w};
#pragma unroll
  for (int j = 0; j < 4; ++j) {
    int c = lane * 4 + j;
    o[j] = (bf16)((vv[j] - mean) * rstd * g[c] + b[c]);
  }
}

// ============== weight transpose + fp32 -> bf16 (W[K][128] -> Wt[128][Kpad])
__global__ void prep_wt(const float* __restrict__ W, bf16* __restrict__ Wt,
                        int Kin, int Kpad) {
  int idx = blockIdx.x * 256 + threadIdx.x;
  if (idx >= 128 * Kpad) return;
  int n = idx / Kpad, k = idx - n * Kpad;
  float v = (k < Kin) ? W[(size_t)k * 128 + n] : 0.0f;
  Wt[idx] = (bf16)v;
}

// ============== node feature builder: [vel(15) | mat_enc(16) | wall(6) | 0pad]
__global__ void node_feat(const int* __restrict__ mat, const float* __restrict__ vel,
                          const float* __restrict__ pos, const float* __restrict__ Wm,
                          const float* __restrict__ bm, bf16* __restrict__ F0) {
  int n = blockIdx.x * blockDim.x + threadIdx.x;
  if (n >= NN) return;
  bf16* o = F0 + (size_t)n * 64;
#pragma unroll
  for (int j = 0; j < 15; ++j) o[j] = (bf16)vel[n * 15 + j];
  int m = mat[n];
#pragma unroll
  for (int j = 0; j < 16; ++j) o[15 + j] = (bf16)(Wm[m * 16 + j] + bm[j]);
#pragma unroll
  for (int d = 0; d < 3; ++d) {
    float pv = pos[n * 3 + d];
    o[31 + 2*d]     = (bf16)fminf(fmaxf(pv, 0.0f), RWALL);
    o[31 + 2*d + 1] = (bf16)fminf(fmaxf(1.0f - pv, 0.0f), RWALL);
  }
#pragma unroll
  for (int j = 37; j < 64; ++j) o[j] = (bf16)0.0f;
}

// ============== edge feature builder: [rel(3) | dist(1) | 0pad to 32]
__global__ void edge_feat(const float* __restrict__ pos, const int* __restrict__ nbr,
                          bf16* __restrict__ G0) {
  int e = blockIdx.x * blockDim.x + threadIdx.x;
  if (e >= NE) return;
  int r = nbr[2*e + 0], s = nbr[2*e + 1];
  float dx = pos[3*r+0] - pos[3*s+0];
  float dy = pos[3*r+1] - pos[3*s+1];
  float dz = pos[3*r+2] - pos[3*s+2];
  float d  = sqrtf(dx*dx + dy*dy + dz*dz);
  bf16* o = G0 + (size_t)e * 32;
  o[0] = (bf16)dx; o[1] = (bf16)dy; o[2] = (bf16)dz; o[3] = (bf16)d;
#pragma unroll
  for (int j = 4; j < 32; ++j) o[j] = (bf16)0.0f;
}

// ============== segment-sum pieces ==============
__global__ void zero_f32(float* __restrict__ p, int n) {
  int i = blockIdx.x * 256 + threadIdx.x;
  if (i < n) p[i] = 0.0f;
}
__global__ void scatter_add(const bf16* __restrict__ edges, const int* __restrict__ nbr,
                            float* __restrict__ agg) {
  int i = blockIdx.x * 256 + threadIdx.x;   // over NE*HD exactly
  int e = i >> 7, c = i & 127;
  float v = (float)edges[i];
  __hip_atomic_fetch_add(&agg[(size_t)nbr[2*e] * HD + c], v,
                         __ATOMIC_RELAXED, __HIP_MEMORY_SCOPE_AGENT);
}
__global__ void pack_agg(const float* __restrict__ agg, bf16* __restrict__ xn) {
  int i = blockIdx.x * 256 + threadIdx.x;   // over NN*HD exactly
  int n = i >> 7, c = i & 127;
  xn[(size_t)n * 256 + c] = (bf16)agg[i];
}

// ============== decoder tail: 128 -> 3, fp32 out ==============
__global__ void dec_tail(const bf16* __restrict__ h, const float* __restrict__ W,
                         const float* __restrict__ b, float* __restrict__ out) {
  int idx = blockIdx.x * 256 + threadIdx.x;
  if (idx >= NN * 3) return;
  int n = idx / 3, j = idx - n * 3;
  float s = b[j];
  const bf16* hr = h + (size_t)n * HD;
#pragma unroll 16
  for (int k = 0; k < HD; ++k) s += (float)hr[k] * W[k * 3 + j];
  out[idx] = s;
}

// =====================================================================
extern "C" void kernel_launch(void* const* d_in, const int* in_sizes, int n_in,
                              void* d_out, int out_size, void* d_ws, size_t ws_size,
                              hipStream_t stream) {
  (void)in_sizes; (void)n_in; (void)out_size; (void)ws_size;
  const int* materials  = (const int*)d_in[0];
  const float* vels     = (const float*)d_in[1];
  const float* pos      = (const float*)d_in[2];
  const int* nbr        = (const int*)d_in[3];
  auto P = [&](int i) { return (const float*)d_in[i]; };
  // params flat order (sorted dict keys):
  // 4..9 decoder, 10..15 edge_enc, 16..17 edge_ln, 18..19 material_enc,
  // 20..25 node_enc, 26..27 node_ln, 28+16*i per proc layer:
  //   +0,+1 edge_ln ; +2..+7 edge_mlp ; +8,+9 node_ln ; +10..+15 node_mlp

  char* ws = (char*)d_ws;
  size_t off = 0;
  auto take = [&](size_t bytes) -> void* {
    off = (off + 255) & ~(size_t)255;
    void* p = ws + off; off += bytes; return p;
  };

  bf16*  XN    = (bf16*) take((size_t)NN * 256 * sizeof(bf16)); // [agg | nodes]
  bf16*  edges = (bf16*) take((size_t)NE * HD  * sizeof(bf16));
  bf16*  h1    = (bf16*) take((size_t)NE * HD  * sizeof(bf16));
  bf16*  h2    = (bf16*) take((size_t)NE * HD  * sizeof(bf16));
  float* Yf    = (float*)take((size_t)NE * HD  * sizeof(float));
  float* agg   = (float*)take((size_t)NN * HD  * sizeof(float));
  bf16*  G0 = (bf16*)Yf;   // E x 32, consumed before Yf is written
  bf16*  F0 = (bf16*)agg;  // N x 64, consumed before agg is used

  auto wt = [&](const float* src, int Kin, int Kpad) -> bf16* {
    bf16* dst = (bf16*)take((size_t)128 * Kpad * sizeof(bf16));
    int total = 128 * Kpad;
    prep_wt<<<(total + 255) / 256, 256, 0, stream>>>(src, dst, Kin, Kpad);
    return dst;
  };

  bf16* w_ne0 = wt(P(20), 37, 64);
  bf16* w_ne1 = wt(P(22), 128, 128);
  bf16* w_ne2 = wt(P(24), 128, 128);
  bf16* w_ee0 = wt(P(10), 4, 32);
  bf16* w_ee1 = wt(P(12), 128, 128);
  bf16* w_ee2 = wt(P(14), 128, 128);
  bf16* w_d0  = wt(P(4), 128, 128);
  bf16* w_d1  = wt(P(6), 128, 128);
  bf16 *w_e0[10], *w_e1[10], *w_e2[10], *w_n0[10], *w_n1[10], *w_n2[10];
  for (int i = 0; i < 10; ++i) {
    int L = 28 + 16 * i;
    w_e0[i] = wt(P(L + 2), 384, 384);
    w_e1[i] = wt(P(L + 4), 128, 128);
    w_e2[i] = wt(P(L + 6), 128, 128);
    w_n0[i] = wt(P(L + 10), 256, 256);
    w_n1[i] = wt(P(L + 12), 128, 128);
    w_n2[i] = wt(P(L + 14), 128, 128);
  }

  auto gemm = [&](const bf16* X, int ldx, int M, int K, const bf16* Wt,
                  const float* bias, int relu, bf16* Ybf, int ldy, int ycol,
                  float* Yfo, int gather) {
    GemmP p{};
    p.X = X ? X : Wt;   // never pass a null pointer into the kernel
    p.ldx = ldx; p.edges = edges; p.xn = XN; p.nbr = nbr;
    p.Wt = Wt; p.bias = bias; p.relu = relu;
    p.Ybf = Ybf; p.ldy = ldy; p.ycol = ycol; p.Yf = Yfo;
    if (gather) {                    // edge-MLP layer 0: M = NE, K = 384
      gemm128<4, 384, true><<<M / 256, 128, 0, stream>>>(p);
    } else if (M % 256 == 0 && M >= NE) {    // big GEMMs: 64 rows/wave
      if (K == 32)       gemm128<4, 32,  false><<<M / 256, 128, 0, stream>>>(p);
      else               gemm128<4, 128, false><<<M / 256, 128, 0, stream>>>(p);
    } else {                                  // node-sized GEMMs: 32 rows/wave
      if (K == 64)       gemm128<2, 64,  false><<<M / 128, 128, 0, stream>>>(p);
      else if (K == 256) gemm128<2, 256, false><<<M / 128, 128, 0, stream>>>(p);
      else               gemm128<2, 128, false><<<M / 128, 128, 0, stream>>>(p);
    }
  };
  auto ln = [&](const float* Y, const float* g, const float* b, bf16* dst,
                int ldy, int ycol, int M) {
    ln_kernel<<<M / 4, 128, 0, stream>>>(Y, g, b, dst, ldy, ycol, M);
  };

  // ---------- node encoder ----------
  node_feat<<<NN / 128, 128, 0, stream>>>(materials, vels, pos, P(18), P(19), F0);
  gemm(F0, 64, NN, 64,  w_ne0, P(21), 1, h1, HD, 0, nullptr, 0);
  gemm(h1, HD, NN, HD,  w_ne1, P(23), 1, h2, HD, 0, nullptr, 0);
  gemm(h2, HD, NN, HD,  w_ne2, P(25), 0, nullptr, 0, 0, Yf, 0);
  ln(Yf, P(26), P(27), XN, 256, 128, NN);

  // ---------- edge encoder ----------
  edge_feat<<<NE / 128, 128, 0, stream>>>(pos, nbr, G0);
  gemm(G0, 32, NE, 32,  w_ee0, P(11), 1, h1, HD, 0, nullptr, 0);
  gemm(h1, HD, NE, HD,  w_ee1, P(13), 1, h2, HD, 0, nullptr, 0);
  gemm(h2, HD, NE, HD,  w_ee2, P(15), 0, nullptr, 0, 0, Yf, 0);
  ln(Yf, P(16), P(17), edges, HD, 0, NE);

  // ---------- 10 processor layers ----------
  for (int i = 0; i < 10; ++i) {
    int L = 28 + 16 * i;
    // edge MLP with fused gather of [edges | nodes[recv] | nodes[send]]
    gemm(nullptr, 0, NE, 384, w_e0[i], P(L + 3), 1, h1, HD, 0, nullptr, 1);
    gemm(h1, HD,  NE, HD,  w_e1[i], P(L + 5), 1, h2, HD, 0, nullptr, 0);
    gemm(h2, HD,  NE, HD,  w_e2[i], P(L + 7), 0, nullptr, 0, 0, Yf, 0);
    ln(Yf, P(L + 0), P(L + 1), edges, HD, 0, NE);
    // segment_sum(edges, recv) -> agg -> XN[:,0:128]
    zero_f32<<<(NN * HD) / 256, 256, 0, stream>>>(agg, NN * HD);
    scatter_add<<<(NE * HD) / 256, 256, 0, stream>>>(edges, nbr, agg);
    pack_agg<<<(NN * HD) / 256, 256, 0, stream>>>(agg, XN);
    // node MLP on [agg | nodes] (stride-256 view of XN)
    gemm(XN, 256, NN, 256, w_n0[i], P(L + 11), 1, h1, HD, 0, nullptr, 0);
    gemm(h1, HD,  NN, HD,  w_n1[i], P(L + 13), 1, h2, HD, 0, nullptr, 0);
    gemm(h2, HD,  NN, HD,  w_n2[i], P(L + 15), 0, nullptr, 0, 0, Yf, 0);
    ln(Yf, P(L + 8), P(L + 9), XN, 256, 128, NN);
  }

  // ---------- decoder ----------
  gemm(XN + 128, 256, NN, HD, w_d0, P(5), 1, h1, HD, 0, nullptr, 0);
  gemm(h1, HD, NN, HD, w_d1, P(7), 1, h2, HD, 0, nullptr, 0);
  dec_tail<<<(NN * 3 + 255) / 256, 256, 0, stream>>>(h2, P(8), P(9), (float*)d_out);
}
